// MidAttnBlock_60601988546748
// MI455X (gfx1250) — matrix operand order repaired and checked by reading, not compiled, not run on hardware
//
#include <hip/hip_runtime.h>
#include <hip/hip_fp16.h>

typedef __attribute__((ext_vector_type(16))) _Float16 v16h;
typedef __attribute__((ext_vector_type(8)))  _Float16 v8h;
typedef __attribute__((ext_vector_type(8)))  float    v8f;
typedef __attribute__((ext_vector_type(4)))  int      v4i;

// Problem constants (from reference)
#define BATCH 8
#define INC   96
#define LEN   2048
#define NHEAD 4
#define NH    32      // BATCH*NHEAD head-batches
#define DQK   48      // per-head q/k channels
#define CPAD  64      // padded q/k channel dim for K=32 WMMA steps
#define DQKV  768
#define DATTN 384

// ---------------------------------------------------------------------------
// CDNA5 async global->LDS (GLOBAL_LOAD_ASYNC_TO_LDS_B128, ASYNCcnt).
// The builtin takes typed v4i pointers: (global addrspace(1) src,
// LDS addrspace(3) dst, imm offset, imm cpol). Guarded: falls back to a
// synchronous copy if the builtin is absent (host pass / older toolchain).
// ---------------------------------------------------------------------------
#if defined(__gfx1250__) && __has_builtin(__builtin_amdgcn_global_load_async_to_lds_b128)
#define HAVE_ASYNC_LDS 1
#else
#define HAVE_ASYNC_LDS 0
#endif

#if HAVE_ASYNC_LDS
#define ASYNC_CP16(g, l)                                                       \
  __builtin_amdgcn_global_load_async_to_lds_b128(                              \
      (__attribute__((address_space(1))) v4i*)(uintptr_t)(g),                  \
      (__attribute__((address_space(3))) v4i*)(uint32_t)(uintptr_t)(l), 0, 0)
#else
#define ASYNC_CP16(g, l) (*(v8h*)(l) = *(const v8h*)(g))
#endif

__device__ __forceinline__ void async_wait_barrier() {
#if HAVE_ASYNC_LDS
#if __has_builtin(__builtin_amdgcn_s_wait_asynccnt)
  __builtin_amdgcn_s_wait_asynccnt(0);
#else
  asm volatile("s_wait_asynccnt 0x0" ::: "memory");
#endif
#endif
  __syncthreads();
}

// ---------------------------------------------------------------------------
// WMMA helpers (CDNA5 16x16x32 f16 -> f32)
// ---------------------------------------------------------------------------
__device__ __forceinline__ v8f wmma_f16(v16h a, v16h b, v8f c) {
  return __builtin_amdgcn_wmma_f32_16x16x32_f16(false, a, false, b, (short)0, c,
                                                false, false);
}

// A operand: A[M,K] row-major with leading dim ld (halves). Lane L holds row
// M=L%16; halves h = K (kb + 16*hi + h).
__device__ __forceinline__ v16h frag_a(const _Float16* __restrict__ base,
                                       int row0, int ld, int kb) {
  int lane = threadIdx.x & 31;
  int m = lane & 15, hi = lane >> 4;
  const _Float16* p0 = base + (size_t)(row0 + m) * ld + kb + 16 * hi;
  v8h a0 = *(const v8h*)p0;
  v8h a1 = *(const v8h*)(p0 + 8);
  v16h r;
#pragma unroll
  for (int i = 0; i < 8; ++i) { r[i] = a0[i]; r[i + 8] = a1[i]; }
  return r;
}

// B operand: B[K,N] accessed column-major via BT (N-major rows of length ld).
// Lane L holds column N=L%16; halves h = K (kb + 16*hi + h): one 32B load.
__device__ __forceinline__ v16h frag_b(const _Float16* __restrict__ baseT,
                                       int col0, int ld, int kb) {
  int lane = threadIdx.x & 31;
  int n = lane & 15, hi = lane >> 4;
  return *(const v16h*)(baseT + (size_t)(col0 + n) * ld + kb + 16 * hi);
}

// ---------------------------------------------------------------------------
// GroupNorm statistics: one block per (batch, group); channels of a group are
// contiguous, so group g's data is x + g*cpg*LEN.
// ---------------------------------------------------------------------------
__global__ __launch_bounds__(256) void gn_stats_kernel(
    const float* __restrict__ x, float* __restrict__ stats, int n_per_group) {
  const float* p = x + (size_t)blockIdx.x * n_per_group;
  float s = 0.f, ss = 0.f;
  for (int i = threadIdx.x; i < n_per_group; i += 256) {
    float v = p[i]; s += v; ss += v * v;
  }
  __shared__ float sh0[256], sh1[256];
  sh0[threadIdx.x] = s; sh1[threadIdx.x] = ss;
  __syncthreads();
  for (int off = 128; off > 0; off >>= 1) {
    if ((int)threadIdx.x < off) {
      sh0[threadIdx.x] += sh0[threadIdx.x + off];
      sh1[threadIdx.x] += sh1[threadIdx.x + off];
    }
    __syncthreads();
  }
  if (threadIdx.x == 0) {
    float m = sh0[0] / n_per_group;
    float var = sh1[0] / n_per_group - m * m;
    stats[2 * blockIdx.x]     = m;
    stats[2 * blockIdx.x + 1] = rsqrtf(var + 1e-5f);
  }
}

// GN1 + SiLU, write transposed f16 (B, L, 96) for the QKV GEMM B-operand.
__global__ void gn1_apply_kernel(const float* __restrict__ x,
                                 const float* __restrict__ stats,
                                 const float* __restrict__ w,
                                 const float* __restrict__ bias,
                                 _Float16* __restrict__ xnT) {
  int idx = blockIdx.x * 256 + threadIdx.x;
  if (idx >= BATCH * INC * LEN) return;
  int l = idx & (LEN - 1);
  int c = (idx / LEN) % INC;
  int b = idx / (INC * LEN);
  int g = b * 32 + c / 3;
  float m = stats[2 * g], r = stats[2 * g + 1];
  float v = (x[idx] - m) * r * w[c] + bias[c];
  v = v / (1.f + __expf(-v));                       // SiLU
  xnT[((size_t)b * LEN + l) * INC + c] = (_Float16)v;
}

__global__ void f32_to_f16_kernel(const float* __restrict__ s,
                                  _Float16* __restrict__ d, int n) {
  int i = blockIdx.x * 256 + threadIdx.x;
  if (i < n) d[i] = (_Float16)s[i];
}

// Zero the channel pad [48,64) of every qT/kT row so padded K-steps add 0.
__global__ void padzero_kernel(_Float16* __restrict__ qT,
                               _Float16* __restrict__ kT) {
  int row = blockIdx.x * 256 + threadIdx.x;
  if (row >= NH * LEN) return;
  v8h z = {};
  *(v8h*)(qT + (size_t)row * CPAD + 48) = z;
  *(v8h*)(qT + (size_t)row * CPAD + 56) = z;
  *(v8h*)(kT + (size_t)row * CPAD + 48) = z;
  *(v8h*)(kT + (size_t)row * CPAD + 56) = z;
}

// ---------------------------------------------------------------------------
// QKV GEMM: h(768,2048) = W(768,96) @ xn(96,2048) + b, per batch.
// One wave per 16x16 output tile, K = 96 = 3 WMMA steps. Results are routed
// straight into attention layouts: qT,kT (n,pos,CPAD) f16 (q pre-scaled),
// v (n,96,LEN) f16.
// ---------------------------------------------------------------------------
__global__ __launch_bounds__(256) void qkv_gemm_kernel(
    const _Float16* __restrict__ w16, const _Float16* __restrict__ xnT,
    const float* __restrict__ bias, const float* __restrict__ scale_p,
    _Float16* __restrict__ qT, _Float16* __restrict__ kT,
    _Float16* __restrict__ vv) {
  int wave = threadIdx.x >> 5, lane = threadIdx.x & 31;
  int tj = blockIdx.x * 8 + wave;   // 0..127 position tile
  int to = blockIdx.y;              // 0..47 out-channel tile
  int b  = blockIdx.z;
  const _Float16* xb = xnT + (size_t)b * LEN * INC;
  v8f acc = {};
#pragma unroll
  for (int kb = 0; kb < 96; kb += 32)
    acc = wmma_f16(frag_a(w16, to * 16, 96, kb), frag_b(xb, tj * 16, 96, kb), acc);

  int m = lane & 15, hi = lane >> 4;
  int j  = tj * 16 + m;             // sequence position
  int o0 = to * 16 + 8 * hi;        // first of 8 consecutive out channels
  float scale = *scale_p;
  if (o0 < 192) {                   // ---- q (scaled)
    int n = b * 4 + o0 / DQK, c0 = o0 % DQK;
    v8h pk;
#pragma unroll
    for (int r = 0; r < 8; ++r) pk[r] = (_Float16)((acc[r] + bias[o0 + r]) * scale);
    *(v8h*)(qT + ((size_t)n * LEN + j) * CPAD + c0) = pk;
  } else if (o0 < 384) {            // ---- k
    int oo = o0 - 192;
    int n = b * 4 + oo / DQK, c0 = oo % DQK;
    v8h pk;
#pragma unroll
    for (int r = 0; r < 8; ++r) pk[r] = (_Float16)(acc[r] + bias[o0 + r]);
    *(v8h*)(kT + ((size_t)n * LEN + j) * CPAD + c0) = pk;
  } else {                          // ---- v
    int oo = o0 - 384;
    int n = b * 4 + oo / INC, c0 = oo % INC;
#pragma unroll
    for (int r = 0; r < 8; ++r)
      vv[((size_t)n * INC + c0 + r) * LEN + j] = (_Float16)(acc[r] + bias[o0 + r]);
  }
}

// ---------------------------------------------------------------------------
// Flash attention. Grid (16, 32): 8 waves/block, one 16-query tile per wave.
// Streams 32-key blocks with DOUBLE-BUFFERED K/V tiles: the async DMA fill of
// block i+1 (GLOBAL_LOAD_ASYNC_TO_LDS_B128, ASYNCcnt) overlaps the 10 WMMAs
// of block i; one s_wait_asynccnt + barrier per iteration.
// S tiles via 2 WMMAs each; online softmax; P fed back as B-operand
// (shfl_xor(16) lane exchange); O accumulated in 6 f32 WMMA tiles.
// ---------------------------------------------------------------------------
__global__ __launch_bounds__(256) void attn_kernel(
    const _Float16* __restrict__ qT, const _Float16* __restrict__ kT,
    const _Float16* __restrict__ vv, float* __restrict__ obuf) {
  int n = blockIdx.y, b = n >> 2, h = n & 3;
  int wave = threadIdx.x >> 5, lane = threadIdx.x & 31;
  int jt = blockIdx.x * 8 + wave;   // query tile 0..127
  int m = lane & 15, hi = lane >> 4;

  __shared__ _Float16 sK[2][32 * CPAD];   // 2 x (32 keys x 64 chans)  = 8 KB
  __shared__ _Float16 sV[2][INC * 32];    // 2 x (96 chans x 32 keys)  = 12 KB

  const _Float16* qn = qT + (size_t)n * LEN * CPAD;
  const _Float16* kn = kT + (size_t)n * LEN * CPAD;
  const _Float16* vn = vv + (size_t)n * INC * LEN;

  // Cooperative tile fill: 256 x 16B chunks for K, 384 x 16B chunks for V
  // (waves 0..3 take the second V pass -> no intra-wave divergence).
  auto fill_tile = [&](int kb0, int buf) {
    int t = threadIdx.x;
    {
      int row = t >> 3, ch = t & 7;
      ASYNC_CP16(kn + (size_t)(kb0 + row) * CPAD + ch * 8,
                 &sK[buf][row * CPAD + ch * 8]);
    }
    for (int idx = t; idx < INC * 4; idx += 256) {
      int r2 = idx >> 2, c2 = idx & 3;
      ASYNC_CP16(vn + (size_t)r2 * LEN + kb0 + c2 * 8,
                 &sV[buf][r2 * 32 + c2 * 8]);
    }
  };

  v16h bq0 = frag_b(qn, jt * 16, CPAD, 0);   // Q is loop-invariant
  v16h bq1 = frag_b(qn, jt * 16, CPAD, 32);

  v8f accs[6];
#pragma unroll
  for (int t = 0; t < 6; ++t) accs[t] = (v8f){};
  float mrun = -1e30f, lrun = 0.f;

  fill_tile(0, 0);
  async_wait_barrier();
  int cur = 0;

  for (int kb0 = 0; kb0 < LEN; kb0 += 32) {
    if (kb0 + 32 < LEN) fill_tile(kb0 + 32, cur ^ 1);   // prefetch next block

    const _Float16* cK = &sK[cur][0];
    const _Float16* cV = &sV[cur][0];

    // S tiles: keys [kb0..kb0+15] and [kb0+16..kb0+31] x 16 queries
    v8f s0 = {}, s1 = {};
    s0 = wmma_f16(frag_a(cK, 0, CPAD, 0),  bq0, s0);
    s0 = wmma_f16(frag_a(cK, 0, CPAD, 32), bq1, s0);
    s1 = wmma_f16(frag_a(cK, 16, CPAD, 0),  bq0, s1);
    s1 = wmma_f16(frag_a(cK, 16, CPAD, 32), bq1, s1);

    // Online softmax over keys (column j split across lanes j and j+16)
    float mx = -1e30f;
#pragma unroll
    for (int r = 0; r < 8; ++r) mx = fmaxf(mx, fmaxf(s0[r], s1[r]));
    mx = fmaxf(mx, __shfl_xor(mx, 16, 32));
    float mnew = fmaxf(mrun, mx);
    float p0[8], p1[8], lsum = 0.f;
#pragma unroll
    for (int r = 0; r < 8; ++r) {
      p0[r] = __expf(s0[r] - mnew);
      p1[r] = __expf(s1[r] - mnew);
      lsum += p0[r] + p1[r];
    }
    lsum += __shfl_xor(lsum, 16, 32);
    float corr = __expf(mrun - mnew);
    lrun = lrun * corr + lsum;
    mrun = mnew;
#pragma unroll
    for (int t = 0; t < 6; ++t)
#pragma unroll
      for (int r = 0; r < 8; ++r) accs[t][r] *= corr;

    // Pack P as B operand: lane hi=0 needs S0 rows 0..15 of column j,
    // lane hi=1 needs S1 rows 0..15 -> exchange halves with lane j^16.
    v16h bp;
#pragma unroll
    for (int r = 0; r < 8; ++r) {
      float x0 = __shfl_xor(p0[r], 16, 32);
      float x1 = __shfl_xor(p1[r], 16, 32);
      bp[r]     = (_Float16)(hi ? x1 : p0[r]);
      bp[r + 8] = (_Float16)(hi ? p1[r] : x0);
    }

    // O += V * P   (6 channel tiles, K = 32 keys)
#pragma unroll
    for (int t = 0; t < 6; ++t)
      accs[t] = wmma_f16(frag_a(cV, t * 16, 32, 0), bp, accs[t]);

    async_wait_barrier();   // next buffer ready; cur buffer free for refill
    cur ^= 1;
  }

  float inv = 1.f / lrun;
  int j = jt * 16 + m;
#pragma unroll
  for (int t = 0; t < 6; ++t)
#pragma unroll
    for (int r = 0; r < 8; ++r) {
      int c = t * 16 + 8 * hi + r;
      obuf[((size_t)b * DATTN + h * INC + c) * LEN + j] = accs[t][r] * inv;
    }
}

// GN2 apply -> transposed f16 (B, L, 384) for proj GEMM B-operand.
__global__ void gn2_apply_kernel(const float* __restrict__ obuf,
                                 const float* __restrict__ stats,
                                 const float* __restrict__ w,
                                 const float* __restrict__ bias,
                                 _Float16* __restrict__ oT) {
  int idx = blockIdx.x * 256 + threadIdx.x;
  if (idx >= BATCH * DATTN * LEN) return;
  int l = idx & (LEN - 1);
  int c = (idx / LEN) % DATTN;
  int b = idx / (DATTN * LEN);
  int g = b * 32 + c / 12;
  float v = (obuf[idx] - stats[2 * g]) * stats[2 * g + 1] * w[c] + bias[c];
  oT[((size_t)b * LEN + l) * DATTN + c] = (_Float16)v;
}

// proj GEMM: p(96,2048) = Wp(96,384) @ on(384,2048) + b, per batch.
__global__ __launch_bounds__(256) void proj_gemm_kernel(
    const _Float16* __restrict__ wp, const _Float16* __restrict__ oT,
    const float* __restrict__ bias, float* __restrict__ pbuf) {
  int wave = threadIdx.x >> 5, lane = threadIdx.x & 31;
  int tj = blockIdx.x * 8 + wave;   // 0..127
  int to = blockIdx.y;              // 0..5
  int b  = blockIdx.z;
  const _Float16* ob = oT + (size_t)b * LEN * DATTN;
  v8f acc = {};
#pragma unroll
  for (int kb = 0; kb < DATTN; kb += 32)
    acc = wmma_f16(frag_a(wp, to * 16, DATTN, kb), frag_b(ob, tj * 16, DATTN, kb), acc);
  int m = lane & 15, hi = lane >> 4;
  int j = tj * 16 + m, o0 = to * 16 + 8 * hi;
#pragma unroll
  for (int r = 0; r < 8; ++r)
    pbuf[((size_t)b * INC + o0 + r) * LEN + j] = acc[r] + bias[o0 + r];
}

// GN3 + SiLU (f32, same layout).
__global__ void gn3_silu_kernel(const float* __restrict__ pbuf,
                                const float* __restrict__ stats,
                                const float* __restrict__ w,
                                const float* __restrict__ bias,
                                float* __restrict__ sbuf) {
  int idx = blockIdx.x * 256 + threadIdx.x;
  if (idx >= BATCH * INC * LEN) return;
  int c = (idx / LEN) % INC;
  int b = idx / (INC * LEN);
  int g = b * 32 + c / 3;
  float v = (pbuf[idx] - stats[2 * g]) * stats[2 * g + 1] * w[c] + bias[c];
  sbuf[idx] = v / (1.f + __expf(-v));
}

// k=3 conv (pad 1) + bias + residual. LDS-tiled input (96 x 130 floats).
__global__ __launch_bounds__(256) void conv3_kernel(
    const float* __restrict__ sbuf, const float* __restrict__ cw,
    const float* __restrict__ cb, const float* __restrict__ x,
    float* __restrict__ out) {
  int b = blockIdx.y;
  int l0 = blockIdx.x * 128;
  __shared__ float st[INC * 130];
  for (int idx = threadIdx.x; idx < INC * 130; idx += 256) {
    int c = idx / 130, col = idx % 130;
    int l = l0 + col - 1;
    st[idx] = (l >= 0 && l < LEN) ? sbuf[((size_t)b * INC + c) * LEN + l] : 0.f;
  }
  __syncthreads();
  for (int idx = threadIdx.x; idx < INC * 128; idx += 256) {
    int o = idx / 128, li = idx % 128;
    float sum = cb[o];
    const float* wr = cw + o * INC * 3;
    const float* sc = st + li;
#pragma unroll 4
    for (int c = 0; c < INC; ++c) {
      sum = fmaf(wr[3 * c],     sc[c * 130],     sum);
      sum = fmaf(wr[3 * c + 1], sc[c * 130 + 1], sum);
      sum = fmaf(wr[3 * c + 2], sc[c * 130 + 2], sum);
    }
    size_t oi = ((size_t)b * INC + o) * LEN + l0 + li;
    out[oi] = x[oi] + sum;
  }
}

// ---------------------------------------------------------------------------
// Workspace layout (bytes). Peak ~56 MB with region reuse across phases.
// ---------------------------------------------------------------------------
#define OFF_ST1    ((size_t)0)                 //  2 KB
#define OFF_ST2    ((size_t)2048)
#define OFF_ST3    ((size_t)4096)
#define OFF_W16Q   ((size_t)8192)              // 147456 B
#define OFF_W16P   ((size_t)155648)            //  73728 B
#define OFF_XNT    ((size_t)262144)            // 3 MB
#define OFF_QT     ((size_t)4  << 20)          // 8 MB
#define OFF_KT     ((size_t)12 << 20)          // 8 MB
#define OFF_V16    ((size_t)20 << 20)          // 12 MB
#define OFF_OBUF   ((size_t)33554432)          // 24 MB (f32)
#define OFF_OT16   OFF_QT                      // reuse q/k region after attn
#define OFF_PBUF   OFF_V16                     // reuse v region after attn
#define OFF_SBUF   ((size_t)27262976)          // within dead v region

extern "C" void kernel_launch(void* const* d_in, const int* in_sizes, int n_in,
                              void* d_out, int out_size, void* d_ws,
                              size_t ws_size, hipStream_t stream) {
  const float* x     = (const float*)d_in[0];
  const float* gn1w  = (const float*)d_in[1];
  const float* gn1b  = (const float*)d_in[2];
  const float* qkvw  = (const float*)d_in[3];
  const float* qkvb  = (const float*)d_in[4];
  const float* scale = (const float*)d_in[5];
  const float* gn2w  = (const float*)d_in[6];
  const float* gn2b  = (const float*)d_in[7];
  const float* projw = (const float*)d_in[8];
  const float* projb = (const float*)d_in[9];
  const float* gn3w  = (const float*)d_in[10];
  const float* gn3b  = (const float*)d_in[11];
  const float* c3w   = (const float*)d_in[12];
  const float* c3b   = (const float*)d_in[13];
  float* out = (float*)d_out;

  char* ws = (char*)d_ws;
  float*    st1  = (float*)(ws + OFF_ST1);
  float*    st2  = (float*)(ws + OFF_ST2);
  float*    st3  = (float*)(ws + OFF_ST3);
  _Float16* w16q = (_Float16*)(ws + OFF_W16Q);
  _Float16* w16p = (_Float16*)(ws + OFF_W16P);
  _Float16* xnT  = (_Float16*)(ws + OFF_XNT);
  _Float16* qT   = (_Float16*)(ws + OFF_QT);
  _Float16* kT   = (_Float16*)(ws + OFF_KT);
  _Float16* v16  = (_Float16*)(ws + OFF_V16);
  float*    obuf = (float*)(ws + OFF_OBUF);
  _Float16* oT   = (_Float16*)(ws + OFF_OT16);
  float*    pbuf = (float*)(ws + OFF_PBUF);
  float*    sbuf = (float*)(ws + OFF_SBUF);

  // 1. GN1 stats + apply (+SiLU) -> xnT f16
  gn_stats_kernel<<<BATCH * 32, 256, 0, stream>>>(x, st1, 3 * LEN);
  gn1_apply_kernel<<<(BATCH * INC * LEN) / 256, 256, 0, stream>>>(x, st1, gn1w, gn1b, xnT);

  // 2. Weight conversion + q/k pad zeroing
  f32_to_f16_kernel<<<(DQKV * INC + 255) / 256, 256, 0, stream>>>(qkvw, w16q, DQKV * INC);
  f32_to_f16_kernel<<<(INC * DATTN + 255) / 256, 256, 0, stream>>>(projw, w16p, INC * DATTN);
  padzero_kernel<<<(NH * LEN) / 256, 256, 0, stream>>>(qT, kT);

  // 3. QKV GEMM (WMMA) -> qT/kT/v
  qkv_gemm_kernel<<<dim3(16, 48, BATCH), 256, 0, stream>>>(w16q, xnT, qkvb, scale,
                                                           qT, kT, v16);

  // 4. Flash attention (WMMA + async LDS double-buffer) -> obuf (B,384,L) f32
  attn_kernel<<<dim3(16, NH), 256, 0, stream>>>(qT, kT, v16, obuf);

  // 5. GN2 stats + apply -> oT f16
  gn_stats_kernel<<<BATCH * 32, 256, 0, stream>>>(obuf, st2, 12 * LEN);
  gn2_apply_kernel<<<(BATCH * DATTN * LEN) / 256, 256, 0, stream>>>(obuf, st2, gn2w, gn2b, oT);

  // 6. proj GEMM (WMMA) -> pbuf f32
  proj_gemm_kernel<<<dim3(16, 6, BATCH), 256, 0, stream>>>(w16p, oT, projb, pbuf);

  // 7. GN3 + SiLU, then k=3 conv + bias + residual
  gn_stats_kernel<<<BATCH * 32, 256, 0, stream>>>(pbuf, st3, 3 * LEN);
  gn3_silu_kernel<<<(BATCH * INC * LEN) / 256, 256, 0, stream>>>(pbuf, st3, gn3w, gn3b, sbuf);
  conv3_kernel<<<dim3(LEN / 128, BATCH), 256, 0, stream>>>(sbuf, c3w, c3b, x, out);
}
